// MultiHeadSelfAttention_90469191123075
// MI455X (gfx1250) — matrix operand order repaired and checked by reading, not compiled, not run on hardware
//
#include <hip/hip_runtime.h>
#include <hip/hip_bf16.h>

#define EMBED 1024
#define HEADS 16
#define HDIM  64
#define SEQ   2048
#define BATCH 2
#define MROWS (BATCH * SEQ) /* 4096 */

typedef __attribute__((ext_vector_type(16))) __bf16 v16bf;
typedef __attribute__((ext_vector_type(8)))  __bf16 v8bf;
typedef __attribute__((ext_vector_type(8)))  float  v8f;
typedef __attribute__((ext_vector_type(4)))  int    v4i;

#if defined(__has_builtin)
#if __has_builtin(__builtin_amdgcn_global_load_async_to_lds_b128)
#define GEMM_ASYNC 1
#endif
#endif
#ifndef GEMM_ASYNC
#define GEMM_ASYNC 0
#endif

static __device__ __forceinline__ v8f vzero8() {
  v8f z;
#pragma unroll
  for (int i = 0; i < 8; ++i) z[i] = 0.0f;
  return z;
}

static __device__ __forceinline__ v8f wmma_bf16(v16bf a, v16bf b, v8f c) {
  // D = A(16x32 bf16) * B(32x16 bf16) + C(16x16 f32)
  return __builtin_amdgcn_wmma_f32_16x16x32_bf16(false, a, false, b,
                                                 (short)0, c, false, false);
}

static __device__ __forceinline__ v16bf concat8(v8bf lo, v8bf hi) {
  return __builtin_shufflevector(lo, hi, 0, 1, 2, 3, 4, 5, 6, 7, 8, 9, 10, 11,
                                 12, 13, 14, 15);
}

// ---- DPP 16-lane row reductions (VALU pipe, no LDS traffic) --------------
// xor1 = quad_perm(1,0,3,2)=0xB1, xor2 = quad_perm(2,3,0,1)=0x4E,
// xor4-equiv = row_half_mirror (0x141), xor8-equiv = row_mirror (0x140).
static __device__ __forceinline__ float dpp_max16(float v) {
  int t;
  t = __builtin_amdgcn_update_dpp(0, __builtin_bit_cast(int, v), 0xB1, 0xF, 0xF, true);
  v = fmaxf(v, __builtin_bit_cast(float, t));
  t = __builtin_amdgcn_update_dpp(0, __builtin_bit_cast(int, v), 0x4E, 0xF, 0xF, true);
  v = fmaxf(v, __builtin_bit_cast(float, t));
  t = __builtin_amdgcn_update_dpp(0, __builtin_bit_cast(int, v), 0x141, 0xF, 0xF, true);
  v = fmaxf(v, __builtin_bit_cast(float, t));
  t = __builtin_amdgcn_update_dpp(0, __builtin_bit_cast(int, v), 0x140, 0xF, 0xF, true);
  v = fmaxf(v, __builtin_bit_cast(float, t));
  return v;
}
static __device__ __forceinline__ float dpp_sum16(float v) {
  int t;
  t = __builtin_amdgcn_update_dpp(0, __builtin_bit_cast(int, v), 0xB1, 0xF, 0xF, true);
  v += __builtin_bit_cast(float, t);
  t = __builtin_amdgcn_update_dpp(0, __builtin_bit_cast(int, v), 0x4E, 0xF, 0xF, true);
  v += __builtin_bit_cast(float, t);
  t = __builtin_amdgcn_update_dpp(0, __builtin_bit_cast(int, v), 0x141, 0xF, 0xF, true);
  v += __builtin_bit_cast(float, t);
  t = __builtin_amdgcn_update_dpp(0, __builtin_bit_cast(int, v), 0x140, 0xF, 0xF, true);
  v += __builtin_bit_cast(float, t);
  return v;
}

// ---- Fragment loaders (contiguous k order: k = 16*(lane/16) + s) ---------------------
static __device__ __forceinline__ v16bf load_a_frag(const __bf16* base, int ld) {
  const int lane = threadIdx.x & 31;
  const int m = lane & 15;
  const int c = (lane >> 4) << 4;
  const __bf16* r = base + (size_t)m * ld;
  v8bf lo = *(const v8bf*)(r + c);
  v8bf hi = *(const v8bf*)(r + c + 8);
  return concat8(lo, hi);
}

static __device__ __forceinline__ v16bf load_b_frag(const __bf16* base, int ld) {
  const int lane = threadIdx.x & 31;
  const int n = lane & 15;
  const int kb = (lane >> 4) << 4;
  return *(const v16bf*)(base + (size_t)n * ld + kb);
}

#if GEMM_ASYNC
// Async DMA: 16B per lane, global -> LDS, tracked by ASYNCcnt.
// Builtin signature (from toolchain diagnostic): (AS1 int4*, AS3 int4*, Ii, Ii)
typedef v4i __attribute__((address_space(1)))* as1_v4i;
typedef v4i __attribute__((address_space(3)))* as3_v4i;
static __device__ __forceinline__ void async_cp16(const __bf16* g, __bf16* l) {
  __builtin_amdgcn_global_load_async_to_lds_b128((as1_v4i)g, (as3_v4i)l, 0, 0);
}
#endif

// ---- fp32 -> bf16 conversion (vectorized x4) -----------------------------
__global__ __launch_bounds__(256) void cvt_f32_bf16(const float* __restrict__ in,
                                                    __bf16* __restrict__ out,
                                                    int n4) {
  int i = blockIdx.x * blockDim.x + threadIdx.x;
  if (i < n4) {
    float4 v = ((const float4*)in)[i];
    __bf16 o[4] = {(__bf16)v.x, (__bf16)v.y, (__bf16)v.z, (__bf16)v.w};
    *(uint64_t*)(out + (size_t)i * 4) = *(const uint64_t*)o;
  }
}

// ---- Tiled WMMA GEMM: Y = A(MxK bf16) * W^T, W is (N x K) bf16 row-major -
// MODE 0: bf16 out [b][h][l][d] (Q/K) ; MODE 1: bf16 out [b][h][d][l] (V^T)
// MODE 2: fp32 out [m][n] + bias (final FC)
#define TILE_K 64
#define KTILES (EMBED / TILE_K) /* 16 */
#define LDSS   72 /* padded bf16 row stride, 144B (16B aligned) */

template <int MODE>
__global__ __launch_bounds__(256) void gemm_wmma(
    const __bf16* __restrict__ A, const __bf16* __restrict__ W,
    __bf16* __restrict__ outb, float* __restrict__ outf,
    const float* __restrict__ bias, int M) {
#if GEMM_ASYNC
  __shared__ __bf16 As[2][128 * LDSS];  // double-buffered
  __shared__ __bf16 Ws[2][128 * LDSS];
#else
  __shared__ __bf16 As[1][128 * LDSS];
  __shared__ __bf16 Ws[1][128 * LDSS];
#endif
  const int K = EMBED;
  const int gridM = M / 128;
  const int bm = blockIdx.x % gridM;
  const int bn = blockIdx.x / gridM;
  const int t = threadIdx.x;
  const int wave = t >> 5, lane = t & 31;
  const int m0 = (wave >> 2) * 64;  // 2x4 wave grid, 64x32 per wave
  const int n0 = (wave & 3) * 32;

  v8f acc[4][2];
#pragma unroll
  for (int i = 0; i < 4; ++i)
#pragma unroll
    for (int j = 0; j < 2; ++j) acc[i][j] = vzero8();

  const int lrow = t >> 1;          // 0..127
  const int coff = (t & 1) << 5;    // 0 or 32 elems (64B per thread per tile)
  const __bf16* ga = &A[(size_t)(bm * 128 + lrow) * K + coff];
  const __bf16* gw = &W[(size_t)(bn * 128 + lrow) * K + coff];
  const int lofs = lrow * LDSS + coff;

#if GEMM_ASYNC
  // Prologue: stage k-tile 0 into buffer 0 (8 async b128 per thread).
#pragma unroll
  for (int q = 0; q < 4; ++q) async_cp16(ga + q * 8, &As[0][lofs + q * 8]);
#pragma unroll
  for (int q = 0; q < 4; ++q) async_cp16(gw + q * 8, &Ws[0][lofs + q * 8]);

  for (int it = 0; it < KTILES; ++it) {
    const int cur = it & 1;
    if (it + 1 < KTILES) {
      const int nxt = cur ^ 1;
      const int kn = (it + 1) * TILE_K;
#pragma unroll
      for (int q = 0; q < 4; ++q)
        async_cp16(ga + kn + q * 8, &As[nxt][lofs + q * 8]);
#pragma unroll
      for (int q = 0; q < 4; ++q)
        async_cp16(gw + kn + q * 8, &Ws[nxt][lofs + q * 8]);
      __asm__ volatile("s_wait_asynccnt 0x8" ::: "memory");  // tile `it` done
    } else {
      __asm__ volatile("s_wait_asynccnt 0x0" ::: "memory");
    }
    __syncthreads();
#else
  for (int it = 0; it < KTILES; ++it) {
    const int cur = 0;
    const int k0 = it * TILE_K;
    *(v16bf*)&As[0][lofs] = *(const v16bf*)(ga + k0);
    *(v16bf*)&As[0][lofs + 16] = *(const v16bf*)(ga + k0 + 16);
    *(v16bf*)&Ws[0][lofs] = *(const v16bf*)(gw + k0);
    *(v16bf*)&Ws[0][lofs + 16] = *(const v16bf*)(gw + k0 + 16);
    __syncthreads();
#endif

#pragma unroll
    for (int ks = 0; ks < 2; ++ks) {  // two 32-deep WMMA steps
      v16bf af[4], bf2[2];
#pragma unroll
      for (int i = 0; i < 4; ++i)
        af[i] = load_a_frag(&As[cur][(m0 + i * 16) * LDSS + ks * 32], LDSS);
#pragma unroll
      for (int j = 0; j < 2; ++j)
        bf2[j] = load_b_frag(&Ws[cur][(n0 + j * 16) * LDSS + ks * 32], LDSS);
#pragma unroll
      for (int i = 0; i < 4; ++i)
#pragma unroll
        for (int j = 0; j < 2; ++j)
          acc[i][j] = wmma_bf16(af[i], bf2[j], acc[i][j]);
    }
    __syncthreads();  // all waves done reading before buffer is re-staged
  }

  // Epilogue: C layout element r -> (m = r + 8*(lane>=16), n = lane&15)
  const int rl = lane & 15, rh = lane >> 4;
#pragma unroll
  for (int i = 0; i < 4; ++i)
#pragma unroll
    for (int j = 0; j < 2; ++j)
#pragma unroll
      for (int r = 0; r < 8; ++r) {
        const int gm = bm * 128 + m0 + i * 16 + r + 8 * rh;
        const int gn = bn * 128 + n0 + j * 16 + rl;
        const float v = acc[i][j][r];
        if constexpr (MODE == 2) {
          outf[(size_t)gm * EMBED + gn] = v + bias[gn];
        } else {
          const int nb = gm >> 11, l = gm & (SEQ - 1);
          const int h = gn >> 6, d = gn & (HDIM - 1);
          size_t idx;
          if constexpr (MODE == 0)
            idx = (((size_t)(nb * HEADS + h) * SEQ) + l) * HDIM + d;
          else
            idx = (((size_t)(nb * HEADS + h) * HDIM) + d) * SEQ + l;
          outb[idx] = (__bf16)v;
        }
      }
  (void)outb; (void)outf; (void)bias;
}

// ---- Flash attention: one wave per 16-row q-tile, 64-wide k-steps --------
// Q,K: [b][h][l][d] bf16 ; Vt: [b][h][d][l] bf16 ; O: [b][l][h*64+d] bf16
#define PSTR 72 /* P row stride (bf16), 144B */
__global__ __launch_bounds__(256) void attn_flash(
    const __bf16* __restrict__ Q, const __bf16* __restrict__ Kb,
    const __bf16* __restrict__ Vt, __bf16* __restrict__ O) {
  __shared__ __bf16 Pls[8 * 16 * PSTR];
  const int wave = threadIdx.x >> 5, lane = threadIdx.x & 31;
  const int task = blockIdx.x * 8 + wave;
  if (task >= BATCH * HEADS * (SEQ / 16)) return;  // wave-uniform
  const int qt = task & (SEQ / 16 - 1);
  const int h = (task >> 7) & (HEADS - 1);
  const int nb = task >> 11;
  // fold 1/sqrt(EMBED) and log2(e) so softmax uses bare v_exp_f32 (exp2)
  const float scale2 = 0.03125f * 1.44269504088896f;

  const __bf16* Qh = Q + (size_t)(nb * HEADS + h) * SEQ * HDIM;
  const __bf16* Kh = Kb + (size_t)(nb * HEADS + h) * SEQ * HDIM;
  const __bf16* Vh = Vt + (size_t)(nb * HEADS + h) * HDIM * SEQ;
  __bf16* P = &Pls[wave * 16 * PSTR];

  const v16bf aq0 = load_a_frag(Qh + (size_t)(qt * 16) * HDIM, HDIM);
  const v16bf aq1 = load_a_frag(Qh + (size_t)(qt * 16) * HDIM + 32, HDIM);

  v8f o[4];
#pragma unroll
  for (int dt = 0; dt < 4; ++dt) o[dt] = vzero8();
  v8f mrow, lrow;
#pragma unroll
  for (int r = 0; r < 8; ++r) { mrow[r] = -1.0e30f; lrow[r] = 0.0f; }

  const int rl = lane & 15, rh = lane >> 4;

  for (int kt = 0; kt < SEQ / 64; ++kt) {
    const int l0 = kt * 64;
    // S tile 16(q) x 64(k'): 4 C tiles, 8 WMMAs (independent across j)
    v8f s[4];
#pragma unroll
    for (int j = 0; j < 4; ++j) s[j] = vzero8();
#pragma unroll
    for (int d = 0; d < 2; ++d) {
      v16bf bk[4];
#pragma unroll
      for (int j = 0; j < 4; ++j)
        bk[j] = load_b_frag(Kh + (size_t)(l0 + j * 16) * HDIM + d * 32, HDIM);
#pragma unroll
      for (int j = 0; j < 4; ++j) s[j] = wmma_bf16(d ? aq1 : aq0, bk[j], s[j]);
    }

    // scale into log2 domain; per-row max over 64 cols via DPP
    v8f mx;
#pragma unroll
    for (int r = 0; r < 8; ++r) {
      s[0][r] *= scale2; s[1][r] *= scale2;
      s[2][r] *= scale2; s[3][r] *= scale2;
      mx[r] = fmaxf(fmaxf(s[0][r], s[1][r]), fmaxf(s[2][r], s[3][r]));
      mx[r] = dpp_max16(mx[r]);
    }

    v8f mnew, f, sm;
#pragma unroll
    for (int r = 0; r < 8; ++r) {
      mnew[r] = fmaxf(mrow[r], mx[r]);
      f[r] = exp2f(mrow[r] - mnew[r]);
      s[0][r] = exp2f(s[0][r] - mnew[r]);
      s[1][r] = exp2f(s[1][r] - mnew[r]);
      s[2][r] = exp2f(s[2][r] - mnew[r]);
      s[3][r] = exp2f(s[3][r] - mnew[r]);
      sm[r] = (s[0][r] + s[1][r]) + (s[2][r] + s[3][r]);
      sm[r] = dpp_sum16(sm[r]);
      lrow[r] = lrow[r] * f[r] + sm[r];
      mrow[r] = mnew[r];
    }
#pragma unroll
    for (int dt = 0; dt < 4; ++dt)
#pragma unroll
      for (int r = 0; r < 8; ++r) o[dt][r] *= f[r];

    // C-layout P (16x64) -> row-major bf16 in per-wave LDS
#pragma unroll
    for (int j = 0; j < 4; ++j)
#pragma unroll
      for (int r = 0; r < 8; ++r)
        P[(r + 8 * rh) * PSTR + j * 16 + rl] = (__bf16)s[j][r];
    __asm__ volatile("s_wait_dscnt 0" ::: "memory");
    v16bf ap0 = load_a_frag(P, PSTR);
    v16bf ap1 = load_a_frag(P + 32, PSTR);

    // O(16x64) += P(16x64) * V(64x64); Vt rows are d, contiguous over l
#pragma unroll
    for (int ks = 0; ks < 2; ++ks) {
      v16bf bv[4];
#pragma unroll
      for (int dt = 0; dt < 4; ++dt)
        bv[dt] = load_b_frag(Vh + (size_t)(dt * 16) * SEQ + l0 + ks * 32, SEQ);
#pragma unroll
      for (int dt = 0; dt < 4; ++dt)
        o[dt] = wmma_bf16(ks ? ap1 : ap0, bv[dt], o[dt]);
    }
  }

  // normalize and write O as [b][l][h*64+d]
#pragma unroll
  for (int dt = 0; dt < 4; ++dt)
#pragma unroll
    for (int r = 0; r < 8; ++r) {
      const int gl = qt * 16 + r + 8 * rh;
      const int e = h * HDIM + dt * 16 + rl;
      O[((size_t)nb * SEQ + gl) * EMBED + e] = (__bf16)(o[dt][r] / lrow[r]);
    }
}

// ---- Host launch ---------------------------------------------------------
extern "C" void kernel_launch(void* const* d_in, const int* in_sizes, int n_in,
                              void* d_out, int out_size, void* d_ws, size_t ws_size,
                              hipStream_t stream) {
  (void)in_sizes; (void)n_in; (void)out_size; (void)ws_size;
  const float* x   = (const float*)d_in[0];
  const float* Wq  = (const float*)d_in[1];
  const float* Wk  = (const float*)d_in[2];
  const float* Wv  = (const float*)d_in[3];
  const float* Wfc = (const float*)d_in[4];
  const float* bfc = (const float*)d_in[5];

  char* p = (char*)d_ws;
  __bf16* xb   = (__bf16*)p; p += (size_t)MROWS * EMBED * 2;
  __bf16* Wqb  = (__bf16*)p; p += (size_t)EMBED * EMBED * 2;
  __bf16* Wkb  = (__bf16*)p; p += (size_t)EMBED * EMBED * 2;
  __bf16* Wvb  = (__bf16*)p; p += (size_t)EMBED * EMBED * 2;
  __bf16* Wfcb = (__bf16*)p; p += (size_t)EMBED * EMBED * 2;
  __bf16* Qb   = (__bf16*)p; p += (size_t)MROWS * EMBED * 2;
  __bf16* Kbf  = (__bf16*)p; p += (size_t)MROWS * EMBED * 2;
  __bf16* Vtb  = (__bf16*)p; p += (size_t)MROWS * EMBED * 2;
  __bf16* Ob   = (__bf16*)p; p += (size_t)MROWS * EMBED * 2;

  {
    int n4 = MROWS * EMBED / 4;
    cvt_f32_bf16<<<(n4 + 255) / 256, 256, 0, stream>>>(x, xb, n4);
    n4 = EMBED * EMBED / 4;
    cvt_f32_bf16<<<(n4 + 255) / 256, 256, 0, stream>>>(Wq, Wqb, n4);
    cvt_f32_bf16<<<(n4 + 255) / 256, 256, 0, stream>>>(Wk, Wkb, n4);
    cvt_f32_bf16<<<(n4 + 255) / 256, 256, 0, stream>>>(Wv, Wvb, n4);
    cvt_f32_bf16<<<(n4 + 255) / 256, 256, 0, stream>>>(Wfc, Wfcb, n4);
  }

  const int nblk = (MROWS / 128) * (EMBED / 128);  // 256
  gemm_wmma<0><<<nblk, 256, 0, stream>>>(xb, Wqb, Qb, nullptr, nullptr, MROWS);
  gemm_wmma<0><<<nblk, 256, 0, stream>>>(xb, Wkb, Kbf, nullptr, nullptr, MROWS);
  gemm_wmma<1><<<nblk, 256, 0, stream>>>(xb, Wvb, Vtb, nullptr, nullptr, MROWS);

  attn_flash<<<(BATCH * HEADS * (SEQ / 16)) / 8, 256, 0, stream>>>(Qb, Kbf, Vtb,
                                                                   Ob);

  gemm_wmma<2><<<nblk, 256, 0, stream>>>(Ob, Wfcb, nullptr, (float*)d_out, bfc,
                                         MROWS);
}
